// STDiT3BlockWithCudaKernel_12171937317143
// MI455X (gfx1250) — 16-bit and 8-bit matrix operand order repaired and checked by reading, compiled, not run on hardware
//
#include <hip/hip_runtime.h>
#include <hip/hip_bf16.h>
#include <stdint.h>

typedef __attribute__((ext_vector_type(16))) _Float16 v16h;
typedef __attribute__((ext_vector_type(8)))  float    v8f;
typedef __attribute__((ext_vector_type(8)))  int      v8i;

constexpr int B_  = 2;
constexpr int T_  = 8;
constexpr int S_  = 256;
constexpr int C_  = 1152;
constexpr int H_  = 16;
constexpr int M_  = 224;
constexpr int HD_ = C_ / H_;
constexpr int DH_ = 4 * C_;
constexpr int N_  = T_ * S_;
constexpr int R_  = B_ * N_;
constexpr int RKV_ = B_ * M_;
constexpr int SIXC_ = 6 * C_;

__global__ void k_mod(const float* __restrict__ sst, const float* __restrict__ t,
                      float* __restrict__ mod, int n) {
  int i = blockIdx.x * blockDim.x + threadIdx.x;
  if (i < n) mod[i] = sst[i % SIXC_] + t[i];
}

__global__ void k_ln_mod(const float* __restrict__ X, const float* __restrict__ mod,
                         float* __restrict__ O, int shiftIdx, int scaleIdx) {
  __shared__ float rs[256], rs2[256];
  const int row = blockIdx.x, tid = threadIdx.x;
  const int b = row / N_;
  const float* xr = X + (size_t)row * C_;
  float s = 0.f, s2 = 0.f;
  for (int c = tid; c < C_; c += 256) { float v = xr[c]; s += v; s2 += v * v; }
  rs[tid] = s; rs2[tid] = s2; __syncthreads();
  for (int k = 128; k > 0; k >>= 1) {
    if (tid < k) { rs[tid] += rs[tid + k]; rs2[tid] += rs2[tid + k]; }
    __syncthreads();
  }
  const float mean = rs[0] / C_;
  const float var  = rs2[0] / C_ - mean * mean;
  const float rstd = rsqrtf(var + 1e-6f);
  const float* mb = mod + (size_t)b * SIXC_;
  float* orow = O + (size_t)row * C_;
  for (int c = tid; c < C_; c += 256)
    orow[c] = (xr[c] - mean) * rstd * (1.f + mb[scaleIdx * C_ + c]) + mb[shiftIdx * C_ + c];
}

__global__ void k_gemm_f16(const float* __restrict__ A, const float* __restrict__ W,
                           const float* __restrict__ bias, float* __restrict__ D,
                           int K, int Nc) {
  __shared__ _Float16 As[64][32];
  __shared__ _Float16 Ws[128][32];
  const int tid  = threadIdx.x;
  const int lane = tid & 31, wave = tid >> 5;
  const int wm = wave >> 1, wn = wave & 1;
  const int r0 = blockIdx.x * 64, c0 = blockIdx.y * 128;

  const int aIdx  = tid * 8;
  const int arow  = aIdx >> 5, akk = aIdx & 31;
  const int wIdx0 = tid * 8;
  const int wrow0 = wIdx0 >> 5, wkk0 = wIdx0 & 31;
  const int wIdx1 = tid * 8 + 2048;
  const int wrow1 = wIdx1 >> 5, wkk1 = wIdx1 & 31;

  const float* aP  = A + (size_t)(r0 + arow)  * K + akk;
  const float* wP0 = W + (size_t)(c0 + wrow0) * K + wkk0;
  const float* wP1 = W + (size_t)(c0 + wrow1) * K + wkk1;

  float4 aR0, aR1, wRa, wRb, wRc, wRd;
  auto loadStage = [&](int kt) {
    aR0 = *(const float4*)(aP  + kt);     aR1 = *(const float4*)(aP  + kt + 4);
    wRa = *(const float4*)(wP0 + kt);     wRb = *(const float4*)(wP0 + kt + 4);
    wRc = *(const float4*)(wP1 + kt);     wRd = *(const float4*)(wP1 + kt + 4);
  };
  auto st4 = [](_Float16* p, const float4& v) {
    p[0] = (_Float16)v.x; p[1] = (_Float16)v.y; p[2] = (_Float16)v.z; p[3] = (_Float16)v.w;
  };

  v8f acc0 = {}, acc1 = {}, acc2 = {}, acc3 = {};
  loadStage(0);
  for (int kt = 0; kt < K; kt += 32) {
    __syncthreads();
    st4(&As[arow][akk], aR0);      st4(&As[arow][akk + 4], aR1);
    st4(&Ws[wrow0][wkk0], wRa);    st4(&Ws[wrow0][wkk0 + 4], wRb);
    st4(&Ws[wrow1][wkk1], wRc);    st4(&Ws[wrow1][wkk1 + 4], wRd);
    __syncthreads();
    if (kt + 32 < K) loadStage(kt + 32);

    v16h a{};
    const _Float16* ar = &As[wm * 16 + (lane & 15)][(lane >> 4) * 16];
    for (int e = 0; e < 16; ++e) { a[e] = ar[e]; }
    const int kg = (lane >> 4) * 16;
    v16h b0{}, b1{}, b2{}, b3{};
    {
      const _Float16* br = &Ws[wn * 64 +  0 + (lane & 15)][kg];
      for (int e = 0; e < 16; ++e) b0[e] = br[e];
    }
    {
      const _Float16* br = &Ws[wn * 64 + 16 + (lane & 15)][kg];
      for (int e = 0; e < 16; ++e) b1[e] = br[e];
    }
    {
      const _Float16* br = &Ws[wn * 64 + 32 + (lane & 15)][kg];
      for (int e = 0; e < 16; ++e) b2[e] = br[e];
    }
    {
      const _Float16* br = &Ws[wn * 64 + 48 + (lane & 15)][kg];
      for (int e = 0; e < 16; ++e) b3[e] = br[e];
    }
    acc0 = __builtin_amdgcn_wmma_f32_16x16x32_f16(false, a, false, b0, (short)0, acc0, false, false);
    acc1 = __builtin_amdgcn_wmma_f32_16x16x32_f16(false, a, false, b1, (short)0, acc1, false, false);
    acc2 = __builtin_amdgcn_wmma_f32_16x16x32_f16(false, a, false, b2, (short)0, acc2, false, false);
    acc3 = __builtin_amdgcn_wmma_f32_16x16x32_f16(false, a, false, b3, (short)0, acc3, false, false);
  }
  const int rbase = r0 + wm * 16 + ((lane >> 4) << 3);
  v8f* accs[4] = {&acc0, &acc1, &acc2, &acc3};
  for (int j = 0; j < 4; ++j) {
    const int col = c0 + wn * 64 + j * 16 + (lane & 15);
    const float bv = bias ? bias[col] : 0.f;
    for (int i = 0; i < 8; ++i)
      D[(size_t)(rbase + i) * Nc + col] = (*accs[j])[i] + bv;
  }
}

__global__ void k_attn(const float* __restrict__ qb, const float* __restrict__ kb,
                       const float* __restrict__ vb, float* __restrict__ ob,
                       int sq, int sk, int sv,
                       long qbs, long kbs, long vbs, long obs,
                       int Lk, float scale) {
  __shared__ float qs[16][HD_];
  __shared__ float sc[16][448];
  __shared__ float red2[16][17];
  const int tid = threadIdx.x;
  const int q0 = blockIdx.x * 16, h = blockIdx.y, bz = blockIdx.z;
  const float* qp = qb + qbs * bz + h * HD_;
  const float* kp = kb + kbs * bz + h * HD_;
  const float* vp = vb + vbs * bz + h * HD_;

  for (int i = tid; i < 16 * HD_; i += 256) {
    int r = i / HD_, d = i - r * HD_;
    qs[r][d] = qp[(size_t)(q0 + r) * sq + d] * scale;
  }
  __syncthreads();
  for (int i = tid; i < 16 * Lk; i += 256) {
    int r = i / Lk, k = i - r * Lk;
    const float* kr = kp + (size_t)k * sk;
    float a = 0.f;
    for (int d = 0; d < HD_; ++d) a += qs[r][d] * kr[d];
    sc[r][k] = a;
  }
  __syncthreads();
  const int rrow = tid >> 4, l16 = tid & 15;
  {
    float m = -3.4e38f;
    for (int k = l16; k < Lk; k += 16) m = fmaxf(m, sc[rrow][k]);
    red2[rrow][l16] = m; __syncthreads();
    if (l16 == 0) {
      float mm = -3.4e38f;
      for (int j = 0; j < 16; ++j) mm = fmaxf(mm, red2[rrow][j]);
      red2[rrow][16] = mm;
    }
    __syncthreads();
    const float rowmax = red2[rrow][16];
    float s = 0.f;
    for (int k = l16; k < Lk; k += 16) {
      float e = expf(sc[rrow][k] - rowmax); sc[rrow][k] = e; s += e;
    }
    __syncthreads();
    red2[rrow][l16] = s; __syncthreads();
    if (l16 == 0) {
      float ss = 0.f;
      for (int j = 0; j < 16; ++j) ss += red2[rrow][j];
      red2[rrow][16] = 1.f / ss;
    }
    __syncthreads();
    const float inv = red2[rrow][16];
    for (int k = l16; k < Lk; k += 16) sc[rrow][k] *= inv;
  }
  __syncthreads();
  for (int i = tid; i < 16 * HD_; i += 256) {
    int r = i / HD_, d = i - r * HD_;
    float a = 0.f;
    for (int k = 0; k < Lk; ++k) a += sc[r][k] * vp[(size_t)k * sv + d];
    ob[obs * bz + (size_t)(q0 + r) * C_ + h * HD_ + d] = a;
  }
}

__global__ void k_quant(const float* __restrict__ X, int8_t* __restrict__ Q,
                        float* __restrict__ S, float* __restrict__ QS, int K) {
  __shared__ float red[256];
  const int row = blockIdx.x, tid = threadIdx.x;
  const float* xr = X + (size_t)row * K;
  float mx = 0.f;
  for (int c = tid; c < K; c += 256) mx = fmaxf(mx, fabsf(xr[c]));
  red[tid] = mx; __syncthreads();
  for (int k = 128; k > 0; k >>= 1) { if (tid < k) red[tid] = fmaxf(red[tid], red[tid + k]); __syncthreads(); }
  const float sval = fmaxf(red[0] / 127.f, 1e-8f);
  const float inv  = 1.f / sval;
  __syncthreads();
  float qsum = 0.f;
  int8_t* qr = Q + (size_t)row * K;
  for (int c = tid; c < K; c += 256) {
    float q = rintf(xr[c] * inv);
    q = fminf(127.f, fmaxf(-127.f, q));
    qr[c] = (int8_t)q; qsum += q;
  }
  red[tid] = qsum; __syncthreads();
  for (int k = 128; k > 0; k >>= 1) { if (tid < k) red[tid] += red[tid + k]; __syncthreads(); }
  if (tid == 0) { S[row] = sval; QS[row] = red[0]; }
}

__device__ __forceinline__ float gelu_tanh(float x) {
  float t = tanhf(0.7978845608f * (x + 0.044715f * x * x * x));
  return 0.5f * x * (1.f + t);
}

template <int MODE>
__global__ void k_gemm_i8(const int8_t* __restrict__ Aq, const int8_t* __restrict__ Wq,
                          const float* __restrict__ as, const float* __restrict__ aqs,
                          const int* __restrict__ zw, const float* __restrict__ sw,
                          const float* __restrict__ bias, const float* __restrict__ res,
                          const float* __restrict__ mod, float* __restrict__ D,
                          int K, int Nc) {
  const int tid  = threadIdx.x;
  const int lane = tid & 31, wave = tid >> 5;
  const int wm = wave >> 1, wn = wave & 1;
  const int r0 = blockIdx.x * 64, c0 = blockIdx.y * 128;
  const int row = r0 + wm * 16 + (lane & 15);

  const int aoff = (lane & 16) ? 16 : 0;
  const int boff = (lane & 16) ? 16 : 0;
  const int8_t* aRow = Aq + (size_t)row * K + aoff;
  const int8_t* wRow0 = Wq + (size_t)(c0 + wn * 64 +  0 + (lane & 15)) * K + boff;
  const int8_t* wRow1 = Wq + (size_t)(c0 + wn * 64 + 16 + (lane & 15)) * K + boff;
  const int8_t* wRow2 = Wq + (size_t)(c0 + wn * 64 + 32 + (lane & 15)) * K + boff;
  const int8_t* wRow3 = Wq + (size_t)(c0 + wn * 64 + 48 + (lane & 15)) * K + boff;

  v8i acc0 = {}, acc1 = {}, acc2 = {}, acc3 = {};
  for (int kt = 0; kt < K; kt += 64) {
    if (kt + 64 < K) {
      __builtin_prefetch(aRow + kt + 64, 0, 1);
      __builtin_prefetch(wRow0 + kt + 64, 0, 1);
    }
    v8i a;
    const int* ap = (const int*)(aRow + kt);
    for (int c = 0; c < 4; ++c) { a[c] = ap[c]; a[4 + c] = ap[8 + c]; }
    auto ldB = [&](const int8_t* base) {
      v8i b;
      const int* wp = (const int*)(base + kt);
      for (int j = 0; j < 4; ++j) { b[j] = wp[j]; b[4 + j] = wp[8 + j]; }
      return b;
    };
    v8i b0 = ldB(wRow0), b1 = ldB(wRow1), b2 = ldB(wRow2), b3 = ldB(wRow3);
    acc0 = __builtin_amdgcn_wmma_i32_16x16x64_iu8(true, a, true, b0, acc0, false, false);
    acc1 = __builtin_amdgcn_wmma_i32_16x16x64_iu8(true, a, true, b1, acc1, false, false);
    acc2 = __builtin_amdgcn_wmma_i32_16x16x64_iu8(true, a, true, b2, acc2, false, false);
    acc3 = __builtin_amdgcn_wmma_i32_16x16x64_iu8(true, a, true, b3, acc3, false, false);
  }
  const int rbase = r0 + wm * 16 + ((lane >> 4) << 3);
  v8i* accs[4] = {&acc0, &acc1, &acc2, &acc3};
  for (int j = 0; j < 4; ++j) {
    const int col = c0 + wn * 64 + j * 16 + (lane & 15);
    const float swc = sw[col], zwc = (float)zw[col], bv = bias[col];
    for (int i = 0; i < 8; ++i) {
      const int r = rbase + i;
      const size_t idx = (size_t)r * Nc + col;
      float v = ((float)(*accs[j])[i] - aqs[r] * zwc) * (as[r] * swc) + bv;
      if (MODE == 0) D[idx] = v;
      else if (MODE == 1) D[idx] = res[idx] + v;
      else if (MODE == 2) D[idx] = gelu_tanh(v);
      else {
        const int b_ = r / N_;
        const float g = mod[(size_t)b_ * SIXC_ + 5 * C_ + col];
        D[idx] = res[idx] + g * v;
      }
    }
  }
}

__global__ void k_resid_gate(float* __restrict__ x, const float* __restrict__ src,
                             const float* __restrict__ mod, int gateIdx, long n) {
  long i = (long)blockIdx.x * blockDim.x + threadIdx.x;
  if (i >= n) return;
  const int col = (int)(i % C_);
  const int b_  = (int)(i / ((long)N_ * C_));
  x[i] += mod[(size_t)b_ * SIXC_ + gateIdx * C_ + col] * src[i];
}

extern "C" void kernel_launch(void* const* d_in, const int* in_sizes, int n_in,
                              void* d_out, int out_size, void* d_ws, size_t ws_size,
                              hipStream_t stream) {
  (void)in_sizes; (void)n_in; (void)out_size; (void)ws_size;
  const float*  x    = (const float*)d_in[0];
  const float*  y    = (const float*)d_in[1];
  const float*  t    = (const float*)d_in[2];
  const float*  sst  = (const float*)d_in[3];
  const float*  Wqkv = (const float*)d_in[4];
  const float*  bqkv = (const float*)d_in[5];
  const float*  Wo   = (const float*)d_in[6];
  const float*  bo   = (const float*)d_in[7];
  const float*  Wkv  = (const float*)d_in[8];
  const float*  bkv  = (const float*)d_in[9];
  const int8_t* qw_q = (const int8_t*)d_in[10];
  const float*  sw_q = (const float*)d_in[11];
  const int*    zw_q = (const int*)d_in[12];
  const float*  b_q  = (const float*)d_in[13];
  const int8_t* qw_cp = (const int8_t*)d_in[14];
  const float*  sw_cp = (const float*)d_in[15];
  const int*    zw_cp = (const int*)d_in[16];
  const float*  b_cp  = (const float*)d_in[17];
  const int8_t* qw_f1 = (const int8_t*)d_in[18];
  const float*  sw_f1 = (const float*)d_in[19];
  const int*    zw_f1 = (const int*)d_in[20];
  const float*  b_f1  = (const float*)d_in[21];
  const int8_t* qw_f2 = (const int8_t*)d_in[22];
  const float*  sw_f2 = (const float*)d_in[23];
  const int*    zw_f2 = (const int*)d_in[24];
  const float*  b_f2  = (const float*)d_in[25];
  float* out = (float*)d_out;

  char* base = (char*)d_ws;
  size_t off = 0;
  auto alloc = [&](size_t bytes) { char* p = base + off; off = (off + bytes + 255) & ~(size_t)255; return p; };
  float*  mod     = (float*)alloc((size_t)B_ * SIXC_ * 4);
  float*  xcur    = (float*)alloc((size_t)R_ * C_ * 4);
  float*  xm      = (float*)alloc((size_t)R_ * C_ * 4);
  float*  qkvbuf  = (float*)alloc((size_t)R_ * 3 * C_ * 4);
  float*  attnbuf = (float*)alloc((size_t)R_ * C_ * 4);
  float*  projbuf = (float*)alloc((size_t)R_ * C_ * 4);
  int8_t* q8      = (int8_t*)alloc((size_t)R_ * DH_);
  float*  qs      = (float*)alloc((size_t)R_ * 4);
  float*  qsum    = (float*)alloc((size_t)R_ * 4);
  float*  qbuf    = (float*)alloc((size_t)R_ * C_ * 4);
  float*  kvbuf   = (float*)alloc((size_t)RKV_ * 2 * C_ * 4);
  float*  h1      = (float*)alloc((size_t)R_ * DH_ * 4);

  const float attn_scale = 1.f / sqrtf((float)HD_);
  const long  nElem = (long)R_ * C_;
  const int   eBlks = (int)((nElem + 255) / 256);

  k_mod<<<(B_ * SIXC_ + 255) / 256, 256, 0, stream>>>(sst, t, mod, B_ * SIXC_);
  hipMemcpyAsync(xcur, x, (size_t)R_ * C_ * 4, hipMemcpyDeviceToDevice, stream);

  k_ln_mod<<<R_, 256, 0, stream>>>(xcur, mod, xm,  0,  1);
  k_gemm_f16<<<dim3(R_ / 64, 3 * C_ / 128), 256, 0, stream>>>(xm, Wqkv, bqkv, qkvbuf, C_, 3 * C_);
  k_attn<<<dim3(S_ / 16, H_, B_ * T_), 256, 0, stream>>>(
      qkvbuf, qkvbuf + C_, qkvbuf + 2 * C_, attnbuf,
      3 * C_, 3 * C_, 3 * C_,
      (long)S_ * 3 * C_, (long)S_ * 3 * C_, (long)S_ * 3 * C_, (long)S_ * C_,
      S_, attn_scale);
  k_gemm_f16<<<dim3(R_ / 64, C_ / 128), 256, 0, stream>>>(attnbuf, Wo, bo, projbuf, C_, C_);
  k_resid_gate<<<eBlks, 256, 0, stream>>>(xcur, projbuf, mod,  2, nElem);

  k_quant<<<R_, 256, 0, stream>>>(xcur, q8, qs, qsum, C_);
  k_gemm_i8<0><<<dim3(R_ / 64, C_ / 128), 256, 0, stream>>>(
      q8, qw_q, qs, qsum, zw_q, sw_q, b_q, nullptr, nullptr, qbuf, C_, C_);
  k_gemm_f16<<<dim3(RKV_ / 64, 2 * C_ / 128), 256, 0, stream>>>(y, Wkv, bkv, kvbuf, C_, 2 * C_);
  k_attn<<<dim3(R_ / 16, H_, 1), 256, 0, stream>>>(
      qbuf, kvbuf, kvbuf + C_, xm,
      C_, 2 * C_, 2 * C_, 0, 0, 0, 0,
      RKV_, attn_scale);
  k_quant<<<R_, 256, 0, stream>>>(xm, q8, qs, qsum, C_);
  k_gemm_i8<1><<<dim3(R_ / 64, C_ / 128), 256, 0, stream>>>(
      q8, qw_cp, qs, qsum, zw_cp, sw_cp, b_cp, xcur, nullptr, xcur, C_, C_);

  k_ln_mod<<<R_, 256, 0, stream>>>(xcur, mod, xm,  3,  4);
  k_quant<<<R_, 256, 0, stream>>>(xm, q8, qs, qsum, C_);
  k_gemm_i8<2><<<dim3(R_ / 64, DH_ / 128), 256, 0, stream>>>(
      q8, qw_f1, qs, qsum, zw_f1, sw_f1, b_f1, nullptr, nullptr, h1, C_, DH_);
  k_quant<<<R_, 256, 0, stream>>>(h1, q8, qs, qsum, DH_);
  k_gemm_i8<3><<<dim3(R_ / 64, C_ / 128), 256, 0, stream>>>(
      q8, qw_f2, qs, qsum, zw_f2, sw_f2, b_f2, xcur, mod, out, DH_, C_);
}
